// LinearAttentionHead_4329327034976
// MI455X (gfx1250) — matrix operand order repaired and checked by reading, not compiled, not run on hardware
//
#include <hip/hip_runtime.h>

// Problem constants (match reference)
#define BB  32
#define SS  4096
#define DDIM 64
#define KCC 256

typedef __attribute__((ext_vector_type(16))) __bf16 v16bf;
typedef __attribute__((ext_vector_type(8)))  float  v8f;

__device__ __forceinline__ __bf16 f2bf(float f) { return (__bf16)f; }

// A-fragment (16x32, 16-bit): lane = M row, element i -> K = i + 16*lh
__device__ __forceinline__ v16bf load_a_frag(const float* rowptr, int lh) {
  const float* p = rowptr + 16 * lh;
  v16bf r;
#pragma unroll
  for (int i = 0; i < 8; ++i) r[i] = f2bf(p[i]);
#pragma unroll
  for (int i = 0; i < 8; ++i) r[i + 8] = f2bf(p[i + 8]);
  return r;
}

// B-fragment (32x16, 16-bit): lane = N col, element i -> K = i + 16*lh ; K contiguous in memory
__device__ __forceinline__ v16bf load_b_contig(const float* colptr, int lh) {
  const float* p = colptr + 16 * lh;
  v16bf r;
#pragma unroll
  for (int i = 0; i < 16; ++i) r[i] = f2bf(p[i]);
  return r;
}

// B-fragment: K strided by `stride` floats in memory
__device__ __forceinline__ v16bf load_b_strided(const float* colptr, int stride, int lh) {
  const float* p = colptr + 16 * lh * stride;
  v16bf r;
#pragma unroll
  for (int i = 0; i < 16; ++i) r[i] = f2bf(p[i * stride]);
  return r;
}

__device__ __forceinline__ v8f wmma_bf16(v16bf a, v16bf b, v8f c) {
  return __builtin_amdgcn_wmma_f32_16x16x32_bf16(false, a, false, b, (short)0, c, false, false);
}

// ---------------------------------------------------------------------------
// Kernel 1: projections  Kp[b] = E_w @ K[b] + E_b,  Vp[b] = F_w @ V[b] + F_b
// GEMM: [KC,S] x [S,D] -> [KC,D].
// One wave per 16(KC) x 32(D) strip: one shared A fragment feeds two
// independent WMMA chains per K-step (fills WMMA hazard slots, halves A loads).
// grid = (KC/16, D/32, 2*B)
// ---------------------------------------------------------------------------
__global__ void __launch_bounds__(32)
proj_kernel(const float* __restrict__ K, const float* __restrict__ V,
            const float* __restrict__ Ew, const float* __restrict__ Eb,
            const float* __restrict__ Fw, const float* __restrict__ Fb,
            float* __restrict__ Kp, float* __restrict__ Vp) {
  const int lane = threadIdx.x & 31;
  const int lr   = lane & 15;   // M or N within tile
  const int lh   = lane >> 4;   // lane half
  const int m0   = blockIdx.x * 16;                 // KC tile
  const int n0   = blockIdx.y * 32;                 // D strip (0 or 32)
  const int b    = blockIdx.z & (BB - 1);
  const int pj   = blockIdx.z >> 5;                 // 0 -> Kp, 1 -> Vp

  const float* W    = pj ? Fw : Ew;
  const float* bias = pj ? Fb : Eb;
  const float* X    = (pj ? V : K) + (size_t)b * SS * DDIM;
  float*       dst  = (pj ? Vp : Kp) + (size_t)b * KCC * DDIM;

  const float* wrow = W + (size_t)(m0 + lr) * SS;   // A row, contiguous K

  v8f acc0 = {};
  v8f acc1 = {};
#pragma unroll 2
  for (int kb = 0; kb < SS; kb += 32) {
    v16bf a  = load_a_frag(wrow + kb, lh);
    v16bf b0 = load_b_strided(X + (size_t)kb * DDIM + n0 + lr,      DDIM, lh);
    v16bf b1 = load_b_strided(X + (size_t)kb * DDIM + n0 + 16 + lr, DDIM, lh);
    acc0 = wmma_bf16(a, b0, acc0);
    acc1 = wmma_bf16(a, b1, acc1);
  }

#pragma unroll
  for (int r = 0; r < 8; ++r) {
    const int m = m0 + r + 8 * lh;
    const float bv = bias[m];
    dst[(size_t)m * DDIM + n0 + lr]      = acc0[r] + bv;
    dst[(size_t)m * DDIM + n0 + 16 + lr] = acc1[r] + bv;
  }
}

// ---------------------------------------------------------------------------
// Kernel 2: P = softmax(mask(Q @ Kp^T / 8)); out = P @ Vp
// 4 waves per block, each wave owns 16 rows of S.  grid = (S/64, B)
// ---------------------------------------------------------------------------
__global__ void __launch_bounds__(128)
attn_kernel(const float* __restrict__ Q, const float* __restrict__ Kp,
            const float* __restrict__ Vp, float* __restrict__ Out) {
  __shared__ float Plds[4][16][KCC];   // 64 KB

  const int lane = threadIdx.x & 31;
  const int lr   = lane & 15;
  const int lh   = lane >> 4;
  const int w    = threadIdx.x >> 5;
  const int b    = blockIdx.y;
  const int row0 = blockIdx.x * 64 + w * 16;

  const float* Qrow = Q  + (size_t)b * SS * DDIM + (size_t)(row0 + lr) * DDIM;
  const float* Kpb  = Kp + (size_t)b * KCC * DDIM;
  const float* Vpb  = Vp + (size_t)b * KCC * DDIM;

  // Q row-tile as two A fragments (K = D = 64)
  v16bf aq0 = load_a_frag(Qrow, lh);
  v16bf aq1 = load_a_frag(Qrow + 32, lh);

  // Scores: 16 tiles of 16x16 over KC, kept in registers (tiles independent ->
  // compiler interleaves their WMMA chains)
  v8f acc[16];
#pragma unroll
  for (int t = 0; t < 16; ++t) {
    const float* krow = Kpb + (size_t)(t * 16 + lr) * DDIM;  // B col = Kp row, contiguous
    v16bf b0 = load_b_contig(krow, lh);
    v16bf b1 = load_b_contig(krow + 32, lh);
    v8f c = {};
    c = wmma_bf16(aq0, b0, c);
    c = wmma_bf16(aq1, b1, c);
    const int j = t * 16 + lr;                   // KC index
#pragma unroll
    for (int r = 0; r < 8; ++r) {
      const int ig = row0 + r + 8 * lh;          // S index
      const float v = c[r] * 0.125f;             // 1/sqrt(64)
      acc[t][r] = (j < ig) ? -1e10f : v;         // keep j >= i
    }
  }

  // Softmax over KC: per-row reduce across 16 tiles + 16 lanes of the half-group
#pragma unroll
  for (int r = 0; r < 8; ++r) {
    float mx = -3.402823466e38f;
#pragma unroll
    for (int t = 0; t < 16; ++t) mx = fmaxf(mx, acc[t][r]);
#pragma unroll
    for (int s = 1; s < 16; s <<= 1) mx = fmaxf(mx, __shfl_xor(mx, s, 32));
    float sum = 0.f;
#pragma unroll
    for (int t = 0; t < 16; ++t) { float e = __expf(acc[t][r] - mx); acc[t][r] = e; sum += e; }
#pragma unroll
    for (int s = 1; s < 16; s <<= 1) sum += __shfl_xor(sum, s, 32);
    const float inv = 1.0f / sum;
#pragma unroll
    for (int t = 0; t < 16; ++t) acc[t][r] *= inv;
  }

  // Spill normalized P to LDS in row-major [16][KC] so it can be re-read A-layout
#pragma unroll
  for (int t = 0; t < 16; ++t)
#pragma unroll
    for (int r = 0; r < 8; ++r)
      Plds[w][r + 8 * lh][t * 16 + lr] = acc[t][r];

  __syncthreads();

  // out = P @ Vp : M=16 rows, N=64, K=KC=256.
  // K-step outer, 4 independent D-tile accumulators inner: each A fragment is
  // read from LDS once and feeds 4 independent WMMA chains.
  v8f oc[4] = {v8f{}, v8f{}, v8f{}, v8f{}};
#pragma unroll
  for (int jb = 0; jb < 8; ++jb) {
    v16bf a = load_a_frag(&Plds[w][lr][jb * 32], lh);
#pragma unroll
    for (int dt = 0; dt < 4; ++dt) {
      v16bf bf = load_b_strided(Vpb + (size_t)(jb * 32) * DDIM + dt * 16 + lr, DDIM, lh);
      oc[dt] = wmma_bf16(a, bf, oc[dt]);
    }
  }

#pragma unroll
  for (int dt = 0; dt < 4; ++dt)
#pragma unroll
    for (int r = 0; r < 8; ++r)
      Out[(size_t)b * SS * DDIM + (size_t)(row0 + r + 8 * lh) * DDIM + dt * 16 + lr] = oc[dt][r];
}

// ---------------------------------------------------------------------------
extern "C" void kernel_launch(void* const* d_in, const int* in_sizes, int n_in,
                              void* d_out, int out_size, void* d_ws, size_t ws_size,
                              hipStream_t stream) {
  (void)in_sizes; (void)n_in; (void)out_size; (void)ws_size;
  const float* Q  = (const float*)d_in[0];
  const float* K  = (const float*)d_in[1];
  const float* V  = (const float*)d_in[2];
  const float* Ew = (const float*)d_in[3];
  const float* Eb = (const float*)d_in[4];
  const float* Fw = (const float*)d_in[5];
  const float* Fb = (const float*)d_in[6];

  float* Kp = (float*)d_ws;                        // [B, KC, D]  (2 MB)
  float* Vp = Kp + (size_t)BB * KCC * DDIM;        // [B, KC, D]  (2 MB)

  proj_kernel<<<dim3(KCC / 16, DDIM / 32, 2 * BB), 32, 0, stream>>>(
      K, V, Ew, Eb, Fw, Fb, Kp, Vp);
  attn_kernel<<<dim3(SS / 64, BB), 128, 0, stream>>>(Q, Kp, Vp, (float*)d_out);
}
